// GraphEdgeFusionAttention_53128745452150
// MI455X (gfx1250) — matrix operand order repaired and checked by reading, not compiled, not run on hardware
//
#include <hip/hip_runtime.h>
#include <stdint.h>

#define NDIM 256
#define HEADS 8
#define EXPH 16
#define SCALEF 0.17677669529663687f   // 32^-0.5

typedef __attribute__((ext_vector_type(16))) __bf16 v16bf;
typedef __attribute__((ext_vector_type(8)))  float  v8f;

union Frag {
  uint32_t u[8];
  uint4    q[2];
  v16bf    v;
};
static_assert(sizeof(Frag) == 32, "frag size");

__device__ __forceinline__ uint32_t f2bf1(float x) {
  uint32_t u = __float_as_uint(x);
  return (u + 0x7FFFu + ((u >> 16) & 1u)) >> 16;   // RNE truncate to bf16
}
__device__ __forceinline__ uint32_t packbf2(float lo, float hi) {
  return f2bf1(lo) | (f2bf1(hi) << 16);
}
__device__ __forceinline__ v8f wmma_bf16(const Frag& a, const Frag& b, v8f c) {
  return __builtin_amdgcn_wmma_f32_16x16x32_bf16(false, a.v, false, b.v,
                                                 (short)0, c, false, false);
}

// ---------------- elementwise utility kernels ----------------
__global__ void k_zero(float* __restrict__ p, int n) {
  int i = blockIdx.x * blockDim.x + threadIdx.x;
  if (i < n) p[i] = 0.f;
}
__global__ void k_cvt(const float* __restrict__ s, uint16_t* __restrict__ d, int n) {
  int i = blockIdx.x * blockDim.x + threadIdx.x;
  if (i < n) d[i] = (uint16_t)f2bf1(s[i]);
}

// ---------------- generic bf16 GEMM: C[M,256] = A[M,256] @ W[256,256]^T ----------------
// one wave32 per 16x16 output tile; K loop of 8 x (16x16x32) WMMA
__global__ void k_gemm_bf16(const uint16_t* __restrict__ A, const uint16_t* __restrict__ W,
                            float* __restrict__ C) {
  const int lane = threadIdx.x;
  const int col = lane & 15, hi = lane >> 4;
  const int m = blockIdx.x * 16 + col;
  const int f = blockIdx.y * 16 + col;
  const uint32_t* arow = (const uint32_t*)A + (size_t)m * (NDIM / 2);
  const uint32_t* wrow = (const uint32_t*)W + (size_t)f * (NDIM / 2);
  v8f acc;
#pragma unroll
  for (int i = 0; i < 8; i++) acc[i] = 0.f;
#pragma unroll
  for (int ks = 0; ks < 8; ks++) {
    Frag a, b;
    int ab = hi * 8 + ks * 16;            // A dwords contiguous
    a.q[0] = *(const uint4*)(arow + ab);
    a.q[1] = *(const uint4*)(arow + ab + 4);
    int bb = hi * 8 + ks * 16;            // B dwords contiguous
    b.q[0] = *(const uint4*)(wrow + bb);
    b.q[1] = *(const uint4*)(wrow + bb + 4);
    acc = wmma_bf16(a, b, acc);
  }
  float* crow = C + (size_t)(blockIdx.x * 16) * NDIM + blockIdx.y * 16 + col;
#pragma unroll
  for (int v = 0; v < 8; v++) crow[(size_t)(v + 8 * hi) * NDIM] = acc[v];
}

// ---------------- edge pass 1: ek GEMM + logits + head MLP + exp-sum scatter ----------------
__global__ void k_edge_logits(const float* __restrict__ edges,
                              const int* __restrict__ dsti, const int* __restrict__ srci,
                              const uint16_t* __restrict__ Wekb,
                              const float* __restrict__ Q, const float* __restrict__ K,
                              const float* __restrict__ Wexp, const float* __restrict__ Wsq,
                              float* __restrict__ logits, float* __restrict__ ssum) {
  __shared__ int s_dst[16], s_src[16];
  __shared__ uint32_t s_eb[16 * 128];       // 16 edges x 256 bf16 (packed pairs)
  __shared__ float s_ek[16][NDIM];
  __shared__ float s_logit[16][HEADS];
  __shared__ float s_wexp[EXPH * HEADS];
  __shared__ float s_wsq[HEADS * EXPH];
  const int lane = threadIdx.x;
  const int col = lane & 15, hi = lane >> 4;
  const size_t e0 = (size_t)blockIdx.x * 16;

  if (lane < 16) { s_dst[lane] = dsti[e0 + lane]; s_src[lane] = srci[e0 + lane]; }
  for (int i = lane; i < EXPH * HEADS; i += 32) { s_wexp[i] = Wexp[i]; s_wsq[i] = Wsq[i]; }
#pragma unroll 4
  for (int t = 0; t < 64; t++) {
    int idx = t * 32 + lane;                // idx = m*128 + kd
    const float2 p = *(const float2*)(edges + (e0 + (size_t)(idx >> 7)) * NDIM + (idx & 127) * 2);
    s_eb[idx] = packbf2(p.x, p.y);
  }
  __syncthreads();

  // ek[16,256] = edges_tile @ Wek^T  via WMMA
  for (int tf = 0; tf < 16; tf++) {
    v8f acc;
#pragma unroll
    for (int i = 0; i < 8; i++) acc[i] = 0.f;
    const uint32_t* wrow = (const uint32_t*)Wekb + (size_t)(tf * 16 + col) * (NDIM / 2);
#pragma unroll
    for (int ks = 0; ks < 8; ks++) {
      Frag a, b;
      int ab = col * 128 + hi * 8 + ks * 16;
      a.q[0] = *(const uint4*)(s_eb + ab);
      a.q[1] = *(const uint4*)(s_eb + ab + 4);
      int bb = hi * 8 + ks * 16;
      b.q[0] = *(const uint4*)(wrow + bb);
      b.q[1] = *(const uint4*)(wrow + bb + 4);
      acc = wmma_bf16(a, b, acc);
    }
#pragma unroll
    for (int v = 0; v < 8; v++) s_ek[v + 8 * hi][tf * 16 + col] = acc[v];
  }
  __syncthreads();

  // raw logits: logit[m][h] = SCALE * sum_f q[dst,f]*(k[src,f]+ek[m,f])
  {
    const int m = col;
    const float* qrow = Q + (size_t)s_dst[m] * NDIM;
    const float* krow = K + (size_t)s_src[m] * NDIM;
#pragma unroll
    for (int h = 0; h < HEADS; h++) {
      float a = 0.f;
#pragma unroll
      for (int t = 0; t < 4; t++) {
        int f0 = h * 32 + hi * 16 + t * 4;
        float4 q4 = *(const float4*)(qrow + f0);
        float4 k4 = *(const float4*)(krow + f0);
        float4 e4 = *(const float4*)(&s_ek[m][f0]);
        a += q4.x * (k4.x + e4.x) + q4.y * (k4.y + e4.y) +
             q4.z * (k4.z + e4.z) + q4.w * (k4.w + e4.w);
      }
      a *= SCALEF;
      a += __shfl_xor(a, 16, 32);
      if (hi == 0) s_logit[m][h] = a;
    }
  }
  __syncthreads();

  // head MLP (expand -> exact GELU -> squeeze), store logits, scatter exp-sum
  if (lane < 16) {
    const int m = lane;
    const int dn = s_dst[m];
    float la[HEADS], o[HEADS];
#pragma unroll
    for (int h = 0; h < HEADS; h++) { la[h] = s_logit[m][h]; o[h] = 0.f; }
#pragma unroll
    for (int j = 0; j < EXPH; j++) {
      float t = 0.f;
#pragma unroll
      for (int h = 0; h < HEADS; h++) t += la[h] * s_wexp[j * HEADS + h];
      float g = 0.5f * t * (1.f + erff(t * 0.70710678118654752f));
#pragma unroll
      for (int h = 0; h < HEADS; h++) o[h] += g * s_wsq[h * EXPH + j];
    }
#pragma unroll
    for (int h = 0; h < HEADS; h++) {
      logits[(e0 + m) * HEADS + h] = o[h];
      __hip_atomic_fetch_add(&ssum[(size_t)dn * HEADS + h], expf(o[h]),
                             __ATOMIC_RELAXED, __HIP_MEMORY_SCOPE_AGENT);
    }
  }
}

// ---------------- edge pass 2: ev GEMM + softmax weights + weighted scatter ----------------
__global__ void k_edge_scatter(const float* __restrict__ edges,
                               const int* __restrict__ dsti, const int* __restrict__ srci,
                               const uint16_t* __restrict__ Wevb,
                               const float* __restrict__ V,
                               const float* __restrict__ logits, const float* __restrict__ ssum,
                               float* __restrict__ attn_out, float* __restrict__ accum) {
  __shared__ int s_dst[16], s_src[16];
  __shared__ uint32_t s_eb[16 * 128];
  __shared__ float s_w[16][HEADS];
  const int lane = threadIdx.x;
  const int col = lane & 15, hi = lane >> 4;
  const size_t e0 = (size_t)blockIdx.x * 16;

  if (lane < 16) { s_dst[lane] = dsti[e0 + lane]; s_src[lane] = srci[e0 + lane]; }
#pragma unroll 4
  for (int t = 0; t < 64; t++) {
    int idx = t * 32 + lane;
    const float2 p = *(const float2*)(edges + (e0 + (size_t)(idx >> 7)) * NDIM + (idx & 127) * 2);
    s_eb[idx] = packbf2(p.x, p.y);
  }
  __syncthreads();
  if (lane < 16) {
    const int m = lane;
    const int dn = s_dst[m];
#pragma unroll
    for (int h = 0; h < HEADS; h++) {
      float w = expf(logits[(e0 + m) * HEADS + h]) /
                (ssum[(size_t)dn * HEADS + h] + 1e-16f);
      s_w[m][h] = w;
      attn_out[(e0 + m) * HEADS + h] = w;
    }
  }
  __syncthreads();

  for (int tf = 0; tf < 16; tf++) {
    v8f acc;
#pragma unroll
    for (int i = 0; i < 8; i++) acc[i] = 0.f;
    const uint32_t* wrow = (const uint32_t*)Wevb + (size_t)(tf * 16 + col) * (NDIM / 2);
#pragma unroll
    for (int ks = 0; ks < 8; ks++) {
      Frag a, b;
      int ab = col * 128 + hi * 8 + ks * 16;
      a.q[0] = *(const uint4*)(s_eb + ab);
      a.q[1] = *(const uint4*)(s_eb + ab + 4);
      int bb = hi * 8 + ks * 16;
      b.q[0] = *(const uint4*)(wrow + bb);
      b.q[1] = *(const uint4*)(wrow + bb + 4);
      acc = wmma_bf16(a, b, acc);
    }
    const int h = tf >> 1;                  // 16 cols per tile, 32 per head
    const int f = tf * 16 + col;
#pragma unroll
    for (int v = 0; v < 8; v++) {
      int m = v + 8 * hi;
      int dn = s_dst[m], sn = s_src[m];
      float val = s_w[m][h] * (V[(size_t)sn * NDIM + f] + acc[v]);
      __hip_atomic_fetch_add(&accum[(size_t)dn * NDIM + f], val,
                             __ATOMIC_RELAXED, __HIP_MEMORY_SCOPE_AGENT);
    }
  }
}

// ---------------- final GEMM with on-the-fly fp32->bf16 A and bias ----------------
__global__ void k_final(const float* __restrict__ Acc, const uint16_t* __restrict__ Woutb,
                        const float* __restrict__ bout, float* __restrict__ Out) {
  const int lane = threadIdx.x, col = lane & 15, hi = lane >> 4;
  const int m = blockIdx.x * 16 + col;
  const int f = blockIdx.y * 16 + col;
  const float* arow = Acc + (size_t)m * NDIM;
  const uint32_t* wrow = (const uint32_t*)Woutb + (size_t)f * (NDIM / 2);
  v8f acc;
#pragma unroll
  for (int i = 0; i < 8; i++) acc[i] = 0.f;
#pragma unroll
  for (int ks = 0; ks < 8; ks++) {
    Frag a, b;
#pragma unroll
    for (int j = 0; j < 8; j++) {
      int dw = j + hi * 8 + ks * 16;
      const float2 p = *(const float2*)(arow + dw * 2);
      a.u[j] = packbf2(p.x, p.y);
    }
    int bb = hi * 8 + ks * 16;
    b.q[0] = *(const uint4*)(wrow + bb);
    b.q[1] = *(const uint4*)(wrow + bb + 4);
    acc = wmma_bf16(a, b, acc);
  }
  const float bias = bout[f];
  float* orow = Out + (size_t)(blockIdx.x * 16) * NDIM + blockIdx.y * 16 + col;
#pragma unroll
  for (int v = 0; v < 8; v++) orow[(size_t)(v + 8 * hi) * NDIM] = acc[v] + bias;
}

extern "C" void kernel_launch(void* const* d_in, const int* in_sizes, int n_in,
                              void* d_out, int out_size, void* d_ws, size_t ws_size,
                              hipStream_t stream) {
  const float* x     = (const float*)d_in[0];
  const float* edges = (const float*)d_in[1];
  const int*   ei    = (const int*)d_in[2];
  const float* Wq    = (const float*)d_in[3];
  const float* Wk    = (const float*)d_in[4];
  const float* Wv    = (const float*)d_in[5];
  const float* Wek   = (const float*)d_in[6];
  const float* Wev   = (const float*)d_in[7];
  const float* Wexp  = (const float*)d_in[8];
  const float* Wsq   = (const float*)d_in[9];
  const float* Wout  = (const float*)d_in[10];
  const float* bout  = (const float*)d_in[11];

  const int N = in_sizes[0] / NDIM;     // 10000
  const int E = in_sizes[2] / 2;        // 320000
  const int* dsti = ei;
  const int* srci = ei + E;

  // workspace carve-up (~57.5 MB)
  char* w = (char*)d_ws;
  uint16_t* xb    = (uint16_t*)w; w += (size_t)N * NDIM * 2;
  uint16_t* wqb   = (uint16_t*)w; w += (size_t)NDIM * NDIM * 2;
  uint16_t* wkb   = (uint16_t*)w; w += (size_t)NDIM * NDIM * 2;
  uint16_t* wvb   = (uint16_t*)w; w += (size_t)NDIM * NDIM * 2;
  uint16_t* wekb  = (uint16_t*)w; w += (size_t)NDIM * NDIM * 2;
  uint16_t* wevb  = (uint16_t*)w; w += (size_t)NDIM * NDIM * 2;
  uint16_t* woutb = (uint16_t*)w; w += (size_t)NDIM * NDIM * 2;
  float* Qw = (float*)w; w += (size_t)N * NDIM * 4;
  float* Kw = (float*)w; w += (size_t)N * NDIM * 4;
  float* Vw = (float*)w; w += (size_t)N * NDIM * 4;
  float* Lg = (float*)w; w += (size_t)E * HEADS * 4;
  float* Sw = (float*)w; w += (size_t)N * HEADS * 4;   // exp-sums (adjacent to Ac)
  float* Ac = (float*)w; w += (size_t)N * NDIM * 4;    // scatter accumulator

  float* outp  = (float*)d_out;
  float* attnp = outp + (size_t)N * NDIM;

  // zero Sw + Ac (contiguous)
  {
    int nz = N * HEADS + N * NDIM;
    k_zero<<<(nz + 255) / 256, 256, 0, stream>>>(Sw, nz);
  }
  // fp32 -> bf16 conversions
  k_cvt<<<((N * NDIM) + 255) / 256, 256, 0, stream>>>(x, xb, N * NDIM);
  const int WN = NDIM * NDIM;
  k_cvt<<<(WN + 255) / 256, 256, 0, stream>>>(Wq,  wqb,  WN);
  k_cvt<<<(WN + 255) / 256, 256, 0, stream>>>(Wk,  wkb,  WN);
  k_cvt<<<(WN + 255) / 256, 256, 0, stream>>>(Wv,  wvb,  WN);
  k_cvt<<<(WN + 255) / 256, 256, 0, stream>>>(Wek, wekb, WN);
  k_cvt<<<(WN + 255) / 256, 256, 0, stream>>>(Wev, wevb, WN);
  k_cvt<<<(WN + 255) / 256, 256, 0, stream>>>(Wout, woutb, WN);

  // node projections
  dim3 gn(N / 16, NDIM / 16), bw(32);
  k_gemm_bf16<<<gn, bw, 0, stream>>>(xb, wqb, Qw);
  k_gemm_bf16<<<gn, bw, 0, stream>>>(xb, wkb, Kw);
  k_gemm_bf16<<<gn, bw, 0, stream>>>(xb, wvb, Vw);

  // edge passes
  k_edge_logits<<<E / 16, 32, 0, stream>>>(edges, dsti, srci, wekb, Qw, Kw,
                                           Wexp, Wsq, Lg, Sw);
  k_edge_scatter<<<E / 16, 32, 0, stream>>>(edges, dsti, srci, wevb, Vw,
                                            Lg, Sw, attnp, Ac);

  // output projection + bias
  k_final<<<dim3(N / 16, NDIM / 16), bw, 0, stream>>>(Ac, woutb, bout, outp);
}
